// HybridAttention_2327872275075
// MI455X (gfx1250) — matrix operand order repaired and checked by reading, not compiled, not run on hardware
//
#include <hip/hip_runtime.h>

typedef _Float16 f16;
typedef __attribute__((ext_vector_type(16))) _Float16 v16h;
typedef __attribute__((ext_vector_type(8)))  float    v8f;

#define Tn  2048
#define En  1024
#define Hn  16
#define Dn  64
#define HDn 1024

// ---------------- WMMA helpers (CDNA5 wave32, 16x16x32 f16 -> f32) ----------------

__device__ __forceinline__ v8f wmma16(v16h a, v16h b, v8f c) {
  // 8 args: (neg_a, A, neg_b, B, c_mod, C, reuse_a, reuse_b)
  return __builtin_amdgcn_wmma_f32_16x16x32_f16(false, a, false, b, (short)0, c, false, false);
}

// A fragment 16x32 (MxK). A pre-offset to (row0,k0); lda in elements.
// lane l: m = l&15, half = l>>4; vgpr j: K = 16*half+2j
__device__ __forceinline__ v16h load_a16(const f16* A, int lda, int lane) {
  const int m = lane & 15, hl = lane >> 4;
  const f16* p = A + (size_t)m * lda;
  v16h a;
#pragma unroll
  for (int j = 0; j < 8; ++j) {
    const int kk = 16 * hl + 2 * j;
    a[2 * j]     = p[kk];
    a[2 * j + 1] = p[kk + 1];
  }
  return a;
}

// B fragment 32x16, element (kk,n) = W[n*ldw + kk]   (W row-major (N,K) -> B = W^T)
__device__ __forceinline__ v16h load_bT(const f16* W, int ldw, int lane) {
  const int n = lane & 15, hl = lane >> 4;
  const f16* p = W + (size_t)n * ldw + 16 * hl;
  v16h b;
#pragma unroll
  for (int j = 0; j < 8; ++j) { b[2 * j] = p[2 * j]; b[2 * j + 1] = p[2 * j + 1]; }
  return b;
}

// B fragment 32x16, element (kk,n) = W[kk*ldw + n]   (W row-major (K,N))
__device__ __forceinline__ v16h load_bN(const f16* W, int ldw, int lane) {
  const int n = lane & 15, hl = lane >> 4;
  const f16* p = W + (size_t)(16 * hl) * ldw + n;
  v16h b;
#pragma unroll
  for (int j = 0; j < 8; ++j) {
    b[2 * j]     = p[(size_t)(2 * j) * ldw];
    b[2 * j + 1] = p[(size_t)(2 * j + 1) * ldw];
  }
  return b;
}

// ---------------- Generic tiled GEMM: C(MxN) = A(MxK) * op(B), f32 accum ----------------
// One wave computes a 16x64 strip. Block = 8 waves = 128 rows x 64 cols.
template <bool TB, typename OutT>
__global__ __launch_bounds__(256) void gemm16(const f16* __restrict__ A,
                                              const f16* __restrict__ B,
                                              OutT* __restrict__ C,
                                              int K, int lda, int ldb, int ldc) {
  const int lane = threadIdx.x & 31, w = threadIdx.x >> 5;
  const int row0 = blockIdx.y * 128 + w * 16;
  const int col0 = blockIdx.x * 64;
  v8f z = {};
  v8f acc[4];
#pragma unroll
  for (int c = 0; c < 4; ++c) acc[c] = z;

  for (int k0 = 0; k0 < K; k0 += 32) {
    const v16h a = load_a16(A + (size_t)row0 * lda + k0, lda, lane);
#pragma unroll
    for (int c = 0; c < 4; ++c) {
      v16h b;
      if (TB) b = load_bT(B + (size_t)(col0 + 16 * c) * ldb + k0, ldb, lane);
      else    b = load_bN(B + (size_t)k0 * ldb + col0 + 16 * c, ldb, lane);
      acc[c] = wmma16(a, b, acc[c]);
    }
  }
  const int ncol = lane & 15, mo = (lane >> 4) * 8;
#pragma unroll
  for (int c = 0; c < 4; ++c)
#pragma unroll
    for (int r = 0; r < 8; ++r)
      C[(size_t)(row0 + mo + r) * ldc + col0 + 16 * c + ncol] = (OutT)acc[c][r];
}

// ---------------- Online-softmax step shared by all three attention branches ------------
// C-layout convention per wave: lane l -> ncol = l&15, half = l>>4; vgpr r -> row r+8*half.
// s0 = logits for keys [k0,k0+16), s1 = keys [k0+16,k0+32), already masked/scaled.
// Pl = this wave's 16x32 f16 LDS tile. Vb pre-offset to (row k0, col0); ldv in elements.
__device__ __forceinline__ void online_step(v8f s0, v8f s1, float* m8, float* l8, v8f* o,
                                            f16* Pl, const f16* Vb, int ldv, int lane) {
  const int ncol = lane & 15, hl = lane >> 4;
  float nm[8], al[8], rs[8], p0[8], p1[8];
#pragma unroll
  for (int r = 0; r < 8; ++r) nm[r] = fmaxf(s0[r], s1[r]);
#pragma unroll
  for (int mk = 1; mk <= 8; mk <<= 1)
#pragma unroll
    for (int r = 0; r < 8; ++r) nm[r] = fmaxf(nm[r], __shfl_xor(nm[r], mk, 32));
#pragma unroll
  for (int r = 0; r < 8; ++r) {
    nm[r] = fmaxf(nm[r], m8[r]);
    al[r] = __expf(m8[r] - nm[r]);
    m8[r] = nm[r];
    p0[r] = __expf(s0[r] - nm[r]);
    p1[r] = __expf(s1[r] - nm[r]);
    rs[r] = p0[r] + p1[r];
  }
#pragma unroll
  for (int mk = 1; mk <= 8; mk <<= 1)
#pragma unroll
    for (int r = 0; r < 8; ++r) rs[r] += __shfl_xor(rs[r], mk, 32);
#pragma unroll
  for (int r = 0; r < 8; ++r) l8[r] = l8[r] * al[r] + rs[r];
#pragma unroll
  for (int c = 0; c < 4; ++c)
#pragma unroll
    for (int r = 0; r < 8; ++r) o[c][r] *= al[r];
  // C-layout -> A-layout relayout via per-wave LDS tile
#pragma unroll
  for (int r = 0; r < 8; ++r) {
    Pl[(r + 8 * hl) * 32 + ncol]      = (f16)p0[r];
    Pl[(r + 8 * hl) * 32 + ncol + 16] = (f16)p1[r];
  }
  asm volatile("s_wait_dscnt 0x0" ::: "memory");  // per-wave LDS fence (waves diverge in trip count)
  const v16h pa = load_a16(Pl, 32, lane);
#pragma unroll
  for (int c = 0; c < 4; ++c) {
    const v16h b = load_bN(Vb + 16 * c, ldv, lane);
    o[c] = wmma16(pa, b, o[c]);
  }
}

__device__ __forceinline__ float gate_w(const float* gate, int h, int i) {
  const float a = gate[h * 3], b = gate[h * 3 + 1], c = gate[h * 3 + 2];
  const float mx = fmaxf(a, fmaxf(b, c));
  const float ea = __expf(a - mx), eb = __expf(b - mx), ec = __expf(c - mx);
  const float v = (i == 0) ? ea : ((i == 1) ? eb : ec);
  return v / (ea + eb + ec);
}

__device__ __forceinline__ void writeback(f16* comb, int q0, int col0, int ldc,
                                          v8f* o, float* l8, float g, int lane, bool accum) {
  const int ncol = lane & 15, hl = lane >> 4;
#pragma unroll
  for (int c = 0; c < 4; ++c)
#pragma unroll
    for (int r = 0; r < 8; ++r) {
      const size_t idx = (size_t)(q0 + r + 8 * hl) * ldc + col0 + 16 * c + ncol;
      const float v = g * o[c][r] / l8[r];
      comb[idx] = (f16)(accum ? (float)comb[idx] + v : v);
    }
}

// ---------------- Branch 1: causal QKV flash attention ----------------
__global__ __launch_bounds__(128) void flash_qkv_kernel(const f16* __restrict__ Q,
                                                        const f16* __restrict__ Kt,
                                                        const f16* __restrict__ V,
                                                        const float* __restrict__ gate,
                                                        f16* __restrict__ comb) {
  __shared__ f16 Pl[4][16 * 32];
  const int lane = threadIdx.x & 31, w = threadIdx.x >> 5;
  const int h = blockIdx.y;
  const int q0 = blockIdx.x * 64 + w * 16;
  const int ncol = lane & 15, hl = lane >> 4;

  const f16* Qh = Q + (size_t)q0 * HDn + h * Dn;
  const v16h qa0 = load_a16(Qh, HDn, lane);
  const v16h qa1 = load_a16(Qh + 32, HDn, lane);

  v8f z = {};
  float m8[8], l8[8];
  v8f o[4];
#pragma unroll
  for (int c = 0; c < 4; ++c) o[c] = z;
#pragma unroll
  for (int r = 0; r < 8; ++r) { m8[r] = -1e30f; l8[r] = 0.f; }

  const int nt = (q0 + 15) / 32 + 1;
  for (int t = 0; t < nt; ++t) {
    const int k0 = t * 32;
    const f16* Kb = Kt + (size_t)k0 * HDn + h * Dn;
    v8f s0 = z, s1 = z;
    s0 = wmma16(qa0, load_bT(Kb, HDn, lane), s0);
    s0 = wmma16(qa1, load_bT(Kb + 32, HDn, lane), s0);
    s1 = wmma16(qa0, load_bT(Kb + (size_t)16 * HDn, HDn, lane), s1);
    s1 = wmma16(qa1, load_bT(Kb + (size_t)16 * HDn + 32, HDn, lane), s1);
#pragma unroll
    for (int r = 0; r < 8; ++r) {
      const int q = q0 + r + 8 * hl;
      s0[r] = (k0 + ncol <= q)      ? s0[r] * 0.125f : -1e30f;
      s1[r] = (k0 + 16 + ncol <= q) ? s1[r] * 0.125f : -1e30f;
    }
    online_step(s0, s1, m8, l8, o, Pl[w], V + (size_t)k0 * HDn + h * Dn, HDn, lane);
  }
  writeback(comb, q0, h * Dn, HDn, o, l8, gate_w(gate, h, 0), lane, false);
}

// ---------------- Branch 2: RRPRAM (per-key learned scores) ----------------
__global__ __launch_bounds__(128) void flash_rrp_kernel(const float* __restrict__ r_arr,
                                                        const f16* __restrict__ RV,
                                                        const float* __restrict__ gate,
                                                        f16* __restrict__ comb) {
  __shared__ f16 Pl[4][16 * 32];
  const int lane = threadIdx.x & 31, w = threadIdx.x >> 5;
  const int h = blockIdx.y;
  const int q0 = blockIdx.x * 64 + w * 16;
  const int ncol = lane & 15, hl = lane >> 4;

  v8f z = {};
  float m8[8], l8[8];
  v8f o[4];
#pragma unroll
  for (int c = 0; c < 4; ++c) o[c] = z;
#pragma unroll
  for (int r = 0; r < 8; ++r) { m8[r] = -1e30f; l8[r] = 0.f; }

  const float* rh = r_arr + (size_t)h * Tn;
  const int nt = (q0 + 15) / 32 + 1;
  for (int t = 0; t < nt; ++t) {
    const int k0 = t * 32;
    const float rk0 = rh[k0 + ncol] * 0.125f;
    const float rk1 = rh[k0 + 16 + ncol] * 0.125f;
    v8f s0, s1;
#pragma unroll
    for (int r = 0; r < 8; ++r) {
      const int q = q0 + r + 8 * hl;
      s0[r] = (k0 + ncol <= q)      ? rk0 : -1e30f;
      s1[r] = (k0 + 16 + ncol <= q) ? rk1 : -1e30f;
    }
    online_step(s0, s1, m8, l8, o, Pl[w], RV + (size_t)k0 * HDn + h * Dn, HDn, lane);
  }
  writeback(comb, q0, h * Dn, HDn, o, l8, gate_w(gate, h, 1), lane, true);
}

// ---------------- Branch 3: Janus (rank-1 logits, shared across heads) ----------------
__global__ __launch_bounds__(128) void flash_jan_kernel(const float* __restrict__ sc,
                                                        const f16* __restrict__ echo,
                                                        const float* __restrict__ gate,
                                                        f16* __restrict__ comb) {
  __shared__ f16 Pl[4][16 * 32];
  const int lane = threadIdx.x & 31, w = threadIdx.x >> 5;
  const int h = blockIdx.y;  // head == 64-column group of echo
  const int q0 = blockIdx.x * 64 + w * 16;
  const int ncol = lane & 15, hl = lane >> 4;

  float sq8[8];
#pragma unroll
  for (int r = 0; r < 8; ++r) sq8[r] = sc[q0 + r + 8 * hl];

  v8f z = {};
  float m8[8], l8[8];
  v8f o[4];
#pragma unroll
  for (int c = 0; c < 4; ++c) o[c] = z;
#pragma unroll
  for (int r = 0; r < 8; ++r) { m8[r] = -1e30f; l8[r] = 0.f; }

  const int nt = (q0 + 15) / 32 + 1;
  for (int t = 0; t < nt; ++t) {
    const int k0 = t * 32;
    const float sk0 = sc[k0 + ncol];
    const float sk1 = sc[k0 + 16 + ncol];
    v8f s0, s1;
#pragma unroll
    for (int r = 0; r < 8; ++r) {
      const int q = q0 + r + 8 * hl;
      s0[r] = (k0 + ncol <= q)      ? sq8[r] * sk0 : -1e30f;
      s1[r] = (k0 + 16 + ncol <= q) ? sq8[r] * sk1 : -1e30f;
    }
    online_step(s0, s1, m8, l8, o, Pl[w], echo + (size_t)k0 * En + h * Dn, En, lane);
  }
  writeback(comb, q0, h * Dn, HDn, o, l8, gate_w(gate, h, 2), lane, true);
}

// ---------------- Small kernels ----------------
__global__ __launch_bounds__(256) void cvt_kernel(const float* __restrict__ s,
                                                  f16* __restrict__ d, int n) {
  const int i = blockIdx.x * 256 + threadIdx.x;
  if (i < n) d[i] = (f16)s[i];
}

// r[h,t] = sum_e x[t,e] * wr[h,e,t]   (streams the 134MB wr tensor)
__global__ __launch_bounds__(256) void rscore_kernel(const f16* __restrict__ xh,
                                                     const float* __restrict__ wr,
                                                     float* __restrict__ r_out) {
  const int h = blockIdx.y;
  const int t = blockIdx.x * 256 + threadIdx.x;
  const float* w = wr + (size_t)h * En * Tn + t;
  const f16* xr = xh + (size_t)t * En;
  float acc = 0.f;
  for (int e = 0; e < En; ++e) acc += (float)xr[e] * w[(size_t)e * Tn];
  r_out[(size_t)h * Tn + t] = acc;
}

// scores[t] = dot(x[t], echo_back[t]) / sqrt(E)
__global__ __launch_bounds__(256) void jscore_kernel(const f16* __restrict__ xh,
                                                     const f16* __restrict__ eb,
                                                     float* __restrict__ sc) {
  const int t = blockIdx.x * 256 + threadIdx.x;
  const f16* a = xh + (size_t)t * En;
  const f16* b = eb + (size_t)t * En;
  float acc = 0.f;
  for (int e = 0; e < En; ++e) acc += (float)a[e] * (float)b[e];
  sc[t] = acc * 0.03125f;  // 1/sqrt(1024)
}

// ---------------- Launcher ----------------
extern "C" void kernel_launch(void* const* d_in, const int* in_sizes, int n_in,
                              void* d_out, int out_size, void* d_ws, size_t ws_size,
                              hipStream_t stream) {
  (void)in_sizes; (void)n_in; (void)out_size; (void)ws_size;
  const float* x    = (const float*)d_in[0];
  const float* wq   = (const float*)d_in[1];
  const float* wk   = (const float*)d_in[2];
  const float* wv   = (const float*)d_in[3];
  const float* wr   = (const float*)d_in[4];
  const float* wvr  = (const float*)d_in[5];
  const float* wj   = (const float*)d_in[6];
  const float* gate = (const float*)d_in[7];
  const float* wo   = (const float*)d_in[8];
  float* out = (float*)d_out;

  size_t off = 0;
  auto take = [&](size_t bytes) -> char* {
    char* p = (char*)d_ws + off;
    off += (bytes + 255) & ~(size_t)255;
    return p;
  };
  f16* xh    = (f16*)take((size_t)Tn * En * 2);
  f16* wqh   = (f16*)take((size_t)En * En * 2);
  f16* wkh   = (f16*)take((size_t)En * En * 2);
  f16* wvh   = (f16*)take((size_t)En * En * 2);
  f16* wvrh  = (f16*)take((size_t)En * En * 2);
  f16* wjh   = (f16*)take((size_t)En * En * 2);
  f16* woh   = (f16*)take((size_t)En * HDn * 2);
  f16* qh    = (f16*)take((size_t)Tn * HDn * 2);
  f16* kh    = (f16*)take((size_t)Tn * HDn * 2);
  f16* vh    = (f16*)take((size_t)Tn * HDn * 2);
  f16* rvh   = (f16*)take((size_t)Tn * HDn * 2);
  f16* echoh = (f16*)take((size_t)Tn * En * 2);
  f16* ebh   = (f16*)take((size_t)Tn * En * 2);
  f16* comb  = (f16*)take((size_t)Tn * HDn * 2);
  float* r_arr = (float*)take((size_t)Hn * Tn * 4);
  float* sc    = (float*)take((size_t)Tn * 4);

  // f32 -> f16 staging
  cvt_kernel<<<(Tn * En) / 256, 256, 0, stream>>>(x, xh, Tn * En);
  cvt_kernel<<<(En * En) / 256, 256, 0, stream>>>(wq, wqh, En * En);
  cvt_kernel<<<(En * En) / 256, 256, 0, stream>>>(wk, wkh, En * En);
  cvt_kernel<<<(En * En) / 256, 256, 0, stream>>>(wv, wvh, En * En);
  cvt_kernel<<<(En * En) / 256, 256, 0, stream>>>(wvr, wvrh, En * En);
  cvt_kernel<<<(En * En) / 256, 256, 0, stream>>>(wj, wjh, En * En);
  cvt_kernel<<<(En * HDn) / 256, 256, 0, stream>>>(wo, woh, En * HDn);

  // Projections: Y = X @ W^T
  const dim3 gg(HDn / 64, Tn / 128);
  gemm16<true, f16><<<gg, 256, 0, stream>>>(xh, wqh, qh, En, En, En, HDn);
  gemm16<true, f16><<<gg, 256, 0, stream>>>(xh, wkh, kh, En, En, En, HDn);
  gemm16<true, f16><<<gg, 256, 0, stream>>>(xh, wvh, vh, En, En, En, HDn);
  gemm16<true, f16><<<gg, 256, 0, stream>>>(xh, wvrh, rvh, En, En, En, HDn);
  // echo = x @ wj^T ; echo_back = echo @ wj
  gemm16<true, f16><<<dim3(En / 64, Tn / 128), 256, 0, stream>>>(xh, wjh, echoh, En, En, En, En);
  gemm16<false, f16><<<dim3(En / 64, Tn / 128), 256, 0, stream>>>(echoh, wjh, ebh, En, En, En, En);

  // Per-position scores
  rscore_kernel<<<dim3(Tn / 256, Hn), 256, 0, stream>>>(xh, wr, r_arr);
  jscore_kernel<<<Tn / 256, 256, 0, stream>>>(xh, ebh, sc);

  // Three gated attention branches accumulate into comb (launch order serializes them)
  const dim3 fg(Tn / 64, Hn);
  flash_qkv_kernel<<<fg, 128, 0, stream>>>(qh, kh, vh, gate, comb);
  flash_rrp_kernel<<<fg, 128, 0, stream>>>(r_arr, rvh, gate, comb);
  flash_jan_kernel<<<fg, 128, 0, stream>>>(sc, echoh, gate, comb);

  // out = comb @ wo^T  (fp32 output)
  gemm16<true, float><<<dim3(En / 64, Tn / 128), 256, 0, stream>>>(comb, woh, out, HDn, HDn, HDn, En);
}
